// AttentionMemoryFusion_31825707663834
// MI455X (gfx1250) — matrix operand order repaired and checked by reading, not compiled, not run on hardware
//
#include <hip/hip_runtime.h>
#include <hip/hip_bf16.h>
#include <stdint.h>

// ---------------------------------------------------------------------------
// Problem constants
// ---------------------------------------------------------------------------
#define B_ROWS 1024
#define D_DIM  512
#define M_ROWS 65536
#define HD_DIM 64

typedef __bf16 bf16_t;
typedef bf16_t v16bf __attribute__((ext_vector_type(16)));
typedef bf16_t v8bf  __attribute__((ext_vector_type(8)));
typedef float  v8f   __attribute__((ext_vector_type(8)));
typedef float  v4f   __attribute__((ext_vector_type(4)));

// sched_group_barrier masks
#define SG_WMMA   0x008
#define SG_VMEMRD 0x020
#define SG_DSRD   0x100

// ---------------------------------------------------------------------------
// WMMA helpers (CDNA5 wave32, V_WMMA_F32_16X16X32_BF16)
// ---------------------------------------------------------------------------
static __device__ __forceinline__ v8f wmma_bf16(v16bf a, v16bf b, v8f c) {
  return __builtin_amdgcn_wmma_f32_16x16x32_bf16(
      /*neg_a=*/false, a, /*neg_b=*/false, b,
      /*c_mod=*/(short)0, c, /*reuse_a=*/false, /*reuse_b=*/false);
}

// A fragment: 16x32 bf16 tile, row-major source. Lane holds row (lane&15).
// Elements 0..7  <-> K = k0 + h*8 + {0..7}
// Elements 8..15 <-> K = k0 + 16 + h*8 + {0..7}
static __device__ __forceinline__ v16bf a_frag_bf16(const bf16_t* row, int k0, int h) {
  v8bf lo = *(const v8bf*)(row + k0 + h * 8);
  v8bf hi = *(const v8bf*)(row + k0 + 16 + h * 8);
  return __builtin_shufflevector(lo, hi, 0, 1, 2, 3, 4, 5, 6, 7,
                                 8, 9, 10, 11, 12, 13, 14, 15);
}

static __device__ __forceinline__ v8bf cvt8(v8f x) {
  v8bf r;
#pragma unroll
  for (int i = 0; i < 8; ++i) r[i] = (bf16_t)x[i];
  return r;
}

// B fragment: 32x16 bf16. Source is B^T row-major; lane holds B^T row (lane&15),
// K = k0 + h*8 + {0..7} and k0 + 16 + h*8 + {0..7}. Two 16-byte loads.
static __device__ __forceinline__ v16bf b_frag(const bf16_t* rowN, int k0, int h) {
  v8bf lo = *(const v8bf*)(rowN + k0 + h * 8);
  v8bf hi = *(const v8bf*)(rowN + k0 + 16 + h * 8);
  return __builtin_shufflevector(lo, hi, 0, 1, 2, 3, 4, 5, 6, 7,
                                 8, 9, 10, 11, 12, 13, 14, 15);
}

// ---------------------------------------------------------------------------
// f32 -> bf16 elementwise convert (vector-of-8)
// ---------------------------------------------------------------------------
__global__ __launch_bounds__(256) void cvt_bf16_kernel(const float* __restrict__ src,
                                                       bf16_t* __restrict__ dst,
                                                       int n8) {
  int i = blockIdx.x * blockDim.x + threadIdx.x;
  if (i < n8) {
    v8f x = *(const v8f*)(src + (size_t)i * 8);
    *(v8bf*)(dst + (size_t)i * 8) = cvt8(x);
  }
}

// ---------------------------------------------------------------------------
// Fused keys+values projection. One block = one 16-row m-tile of `memory`.
// The f32 A-tile is cooperatively staged (and converted to bf16) in LDS once,
// so the 128MB memory matrix is read exactly once from HBM.
//   keys[M,64]  = bf16(mem @ Wk^T + bk)      (row-major: scores B-operand)
//   Vt[512,M]   = bf16((mem @ Wv^T + bv)^T)  (attended B-operand)
// Block = 128 threads = 4 waves; wave w owns values cols [w*128, w*128+128)
// and keys n-tile w. K-loop is software-pipelined (registers double-buffered)
// and the schedule is pinned with sched_group_barrier: each WMMA of step k is
// braided with two VMEM loads of step k+1.
// ---------------------------------------------------------------------------
__global__ __launch_bounds__(128) void proj_kv_kernel(const float* __restrict__ memory,
                                                      const bf16_t* __restrict__ Wk,
                                                      const float* __restrict__ bk,
                                                      const bf16_t* __restrict__ Wv,
                                                      const float* __restrict__ bv,
                                                      bf16_t* __restrict__ keys,
                                                      bf16_t* __restrict__ vt) {
  __shared__ bf16_t atile[16 * D_DIM];  // 16KB

  const int lane = threadIdx.x & 31;
  const int wv = threadIdx.x >> 5;  // 0..3
  const int ln = lane & 15;
  const int h = lane >> 4;
  const int m0 = blockIdx.x * 16;
  const int d0 = wv * 128;

  // ---- cooperative stage: 16x512 f32 -> bf16 in LDS ----
  const float* asrc = memory + (size_t)m0 * D_DIM;
#pragma unroll
  for (int it = 0; it < (16 * D_DIM / 8) / 128; ++it) {
    int i = it * 128 + threadIdx.x;
    v8f x = *(const v8f*)(asrc + (size_t)i * 8);
    *(v8bf*)(atile + (size_t)i * 8) = cvt8(x);
  }
  __syncthreads();

  v8f acc_v[8];
#pragma unroll
  for (int nt = 0; nt < 8; ++nt) {
    float b = bv[d0 + nt * 16 + ln];
#pragma unroll
    for (int g = 0; g < 8; ++g) acc_v[nt][g] = b;
  }
  v8f acc_k;
  {
    float b = bk[wv * 16 + ln];
#pragma unroll
    for (int g = 0; g < 8; ++g) acc_k[g] = b;
  }

  const bf16_t* arow = atile + (size_t)ln * D_DIM;
  const bf16_t* wkrow = Wk + (size_t)(wv * 16 + ln) * D_DIM;

  // ---- prologue: fragments for k0 = 0 ----
  v16bf a_cur = a_frag_bf16(arow, 0, h);
  v16bf bk_cur = b_frag(wkrow, 0, h);
  v16bf bw_cur[8];
#pragma unroll
  for (int nt = 0; nt < 8; ++nt)
    bw_cur[nt] = b_frag(Wv + (size_t)(d0 + nt * 16 + ln) * D_DIM, 0, h);

#pragma unroll 4
  for (int k0 = 0; k0 < D_DIM; k0 += 32) {
    const int kn = (k0 + 32 < D_DIM) ? (k0 + 32) : 0;  // last reload is dead
    // issue next-step loads first
    v16bf a_nxt = a_frag_bf16(arow, kn, h);
    v16bf bk_nxt = b_frag(wkrow, kn, h);
    v16bf bw_nxt[8];
#pragma unroll
    for (int nt = 0; nt < 8; ++nt)
      bw_nxt[nt] = b_frag(Wv + (size_t)(d0 + nt * 16 + ln) * D_DIM, kn, h);

    // WMMA chain on current fragments
    acc_k = wmma_bf16(a_cur, bk_cur, acc_k);
#pragma unroll
    for (int nt = 0; nt < 8; ++nt) acc_v[nt] = wmma_bf16(a_cur, bw_cur[nt], acc_v[nt]);

    a_cur = a_nxt;
    bk_cur = bk_nxt;
#pragma unroll
    for (int nt = 0; nt < 8; ++nt) bw_cur[nt] = bw_nxt[nt];

    // pin the schedule: braid 9 WMMAs (step k) with 18 VMEM reads (step k+1),
    // then the 2 DS reads for the next A fragment
#pragma unroll
    for (int u = 0; u < 9; ++u) {
      __builtin_amdgcn_sched_group_barrier(SG_WMMA, 1, 0);
      __builtin_amdgcn_sched_group_barrier(SG_VMEMRD, 2, 0);
    }
    __builtin_amdgcn_sched_group_barrier(SG_DSRD, 2, 0);
  }

  // keys n-tile (strided bf16 stores)
#pragma unroll
  for (int g = 0; g < 8; ++g) {
    int row = m0 + h * 8 + g;
    keys[(size_t)row * HD_DIM + wv * 16 + ln] = (bf16_t)acc_k[g];
  }
  // values: transposed store is contiguous (one 16B store per n-tile)
#pragma unroll
  for (int nt = 0; nt < 8; ++nt) {
    int d = d0 + nt * 16 + ln;
    *(v8bf*)(vt + (size_t)d * M_ROWS + m0 + h * 8) = cvt8(acc_v[nt]);
  }
}

// ---------------------------------------------------------------------------
// Query projection: q[1024,64] = bf16(cf_bf @ Wq^T + bq). Wave -> 16x64.
// ---------------------------------------------------------------------------
__global__ __launch_bounds__(256) void proj_q_kernel(const bf16_t* __restrict__ A,
                                                     const bf16_t* __restrict__ W,
                                                     const float* __restrict__ bias,
                                                     bf16_t* __restrict__ out) {
  const int lane = threadIdx.x & 31;
  const int wv = threadIdx.x >> 5;
  const int ln = lane & 15;
  const int h = lane >> 4;
  const int m0 = (blockIdx.x * 8 + wv) * 16;

  v8f acc[4];
#pragma unroll
  for (int nt = 0; nt < 4; ++nt) {
    float b = bias[nt * 16 + ln];
#pragma unroll
    for (int g = 0; g < 8; ++g) acc[nt][g] = b;
  }

  const bf16_t* arow = A + (size_t)(m0 + ln) * D_DIM;
#pragma unroll 4
  for (int k0 = 0; k0 < D_DIM; k0 += 32) {
    v16bf a = a_frag_bf16(arow, k0, h);
    v16bf bw[4];
#pragma unroll
    for (int nt = 0; nt < 4; ++nt)
      bw[nt] = b_frag(W + (size_t)(nt * 16 + ln) * D_DIM, k0, h);
#pragma unroll
    for (int nt = 0; nt < 4; ++nt) acc[nt] = wmma_bf16(a, bw[nt], acc[nt]);
  }

#pragma unroll
  for (int nt = 0; nt < 4; ++nt)
#pragma unroll
    for (int g = 0; g < 8; ++g)
      out[(size_t)(m0 + h * 8 + g) * HD_DIM + nt * 16 + ln] = (bf16_t)acc[nt][g];
}

// ---------------------------------------------------------------------------
// Softmax pass 1: per-row max & sum(exp(score)) over all M.
// Block = 8 waves sharing 64 query rows (4 row-tiles); each wave strides over
// m-tiles; keys fragments are reused across the 4 row-tiles and double-
// buffered across m-tiles (loads hide under the online-softmax VALU work).
// ---------------------------------------------------------------------------
__global__ __launch_bounds__(256) void stats_kernel(const bf16_t* __restrict__ qbf,
                                                    const bf16_t* __restrict__ keys,
                                                    float* __restrict__ stats) {
  __shared__ float smax[8][64];
  __shared__ float ssum[8][64];

  const int lane = threadIdx.x & 31;
  const int wv = threadIdx.x >> 5;
  const int ln = lane & 15;
  const int h = lane >> 4;
  const int qb = blockIdx.x * 64;
  const float scale = 0.125f;  // 1/sqrt(64)

  v16bf aq[4][2];
#pragma unroll
  for (int rt = 0; rt < 4; ++rt) {
    const bf16_t* qrow = qbf + (size_t)(qb + rt * 16 + ln) * HD_DIM;
    aq[rt][0] = a_frag_bf16(qrow, 0, h);
    aq[rt][1] = a_frag_bf16(qrow, 32, h);
  }

  float rm[4][8], rs[4][8];
#pragma unroll
  for (int rt = 0; rt < 4; ++rt)
#pragma unroll
    for (int g = 0; g < 8; ++g) { rm[rt][g] = -3.0e38f; rs[rt][g] = 0.f; }

  const int mt_end = M_ROWS / 16;
  const bf16_t* krow = keys + (size_t)(wv * 16 + ln) * HD_DIM;
  v16bf kb0 = b_frag(krow, 0, h);
  v16bf kb1 = b_frag(krow, 32, h);

  for (int mt = wv; mt < mt_end; mt += 8) {
    // preload next m-tile's keys fragments
    const int mtn = (mt + 8 < mt_end) ? (mt + 8) : wv;
    const bf16_t* krown = keys + (size_t)(mtn * 16 + ln) * HD_DIM;
    __builtin_prefetch(krown + 8 * 16 * HD_DIM, 0, 0);
    v16bf kb0n = b_frag(krown, 0, h);
    v16bf kb1n = b_frag(krown, 32, h);

#pragma unroll
    for (int rt = 0; rt < 4; ++rt) {
      v8f c = {0.f, 0.f, 0.f, 0.f, 0.f, 0.f, 0.f, 0.f};
      c = wmma_bf16(aq[rt][0], kb0, c);
      c = wmma_bf16(aq[rt][1], kb1, c);
#pragma unroll
      for (int g = 0; g < 8; ++g) {
        float s = c[g] * scale;
        float nm = fmaxf(rm[rt][g], s);
        rs[rt][g] = rs[rt][g] * __expf(rm[rt][g] - nm) + __expf(s - nm);
        rm[rt][g] = nm;
      }
    }
    kb0 = kb0n;
    kb1 = kb1n;

    // braid: 8 WMMAs with the 4 next-tile VMEM reads; exp VALU fills the rest
#pragma unroll
    for (int u = 0; u < 4; ++u) {
      __builtin_amdgcn_sched_group_barrier(SG_WMMA, 2, 0);
      __builtin_amdgcn_sched_group_barrier(SG_VMEMRD, 1, 0);
    }
  }

  // reduce across the 16 lanes of each half-wave (memory-column direction)
#pragma unroll
  for (int off = 1; off < 16; off <<= 1) {
#pragma unroll
    for (int rt = 0; rt < 4; ++rt)
#pragma unroll
      for (int g = 0; g < 8; ++g) {
        float om = __shfl_xor(rm[rt][g], off, 32);
        float os = __shfl_xor(rs[rt][g], off, 32);
        float nm = fmaxf(rm[rt][g], om);
        rs[rt][g] = rs[rt][g] * __expf(rm[rt][g] - nm) + os * __expf(om - nm);
        rm[rt][g] = nm;
      }
  }

  if (ln == 0) {
#pragma unroll
    for (int rt = 0; rt < 4; ++rt)
#pragma unroll
      for (int g = 0; g < 8; ++g) {
        smax[wv][rt * 16 + h * 8 + g] = rm[rt][g];
        ssum[wv][rt * 16 + h * 8 + g] = rs[rt][g];
      }
  }
  __syncthreads();

  if (threadIdx.x < 64) {
    int row = threadIdx.x;
    float m = -3.0e38f, s = 0.f;
#pragma unroll
    for (int w = 0; w < 8; ++w) {
      float om = smax[w][row], os = ssum[w][row];
      float nm = fmaxf(m, om);
      s = s * __expf(m - nm) + os * __expf(om - nm);
      m = nm;
    }
    stats[(size_t)(qb + row) * 2 + 0] = m;
    stats[(size_t)(qb + row) * 2 + 1] = s;
  }
}

// ---------------------------------------------------------------------------
// Pass 2: attended = softmax_weights @ values, weights generated on the fly.
// Block = 4 waves sharing 64 query rows (4 row-tiles); each wave owns 64
// D-columns and one score m-subtile per K-chunk of 64. Every keys/Vt fragment
// is reused across 4 row-tiles; score tiles transpose C->A through LDS.
// Schedules in both barrier-delimited regions are pinned with
// sched_group_barrier so loads braid with the WMMA chains.
// ---------------------------------------------------------------------------
__global__ __launch_bounds__(128) void attended_kernel(const bf16_t* __restrict__ qbf,
                                                       const bf16_t* __restrict__ keys,
                                                       const bf16_t* __restrict__ vt,
                                                       const float* __restrict__ stats,
                                                       bf16_t* __restrict__ att_bf) {
  __shared__ bf16_t wtile[64][64];  // 8KB

  const int lane = threadIdx.x & 31;
  const int wv = threadIdx.x >> 5;  // 0..3
  const int ln = lane & 15;
  const int h = lane >> 4;
  const int qb = blockIdx.x * 64;
  const int d_base = blockIdx.y * 256 + wv * 64;
  const float scale = 0.125f;

  v16bf aq[4][2];
#pragma unroll
  for (int rt = 0; rt < 4; ++rt) {
    const bf16_t* qrow = qbf + (size_t)(qb + rt * 16 + ln) * HD_DIM;
    aq[rt][0] = a_frag_bf16(qrow, 0, h);
    aq[rt][1] = a_frag_bf16(qrow, 32, h);
  }

  float rmax[4][8], rinv[4][8];
#pragma unroll
  for (int rt = 0; rt < 4; ++rt)
#pragma unroll
    for (int g = 0; g < 8; ++g) {
      int row = qb + rt * 16 + h * 8 + g;
      rmax[rt][g] = stats[(size_t)row * 2 + 0];
      rinv[rt][g] = 1.0f / stats[(size_t)row * 2 + 1];
    }

  v8f acc[4][4];
#pragma unroll
  for (int rt = 0; rt < 4; ++rt)
#pragma unroll
    for (int nt = 0; nt < 4; ++nt)
#pragma unroll
      for (int g = 0; g < 8; ++g) acc[rt][nt][g] = 0.f;

  // per-wave Vt row bases (d column fixed per nt)
  const bf16_t* vrow[4];
#pragma unroll
  for (int nt = 0; nt < 4; ++nt)
    vrow[nt] = vt + (size_t)(d_base + nt * 16 + ln) * M_ROWS;

  const int kc_end = M_ROWS / 16;
  const bf16_t* krow = keys + (size_t)(wv * 16 + ln) * HD_DIM;
  v16bf kb0 = b_frag(krow, 0, h);
  v16bf kb1 = b_frag(krow, 32, h);

  for (int kc = 0; kc < kc_end; kc += 4) {
    // ---- scores for my m-subtile, all 4 row-tiles ----
#pragma unroll
    for (int rt = 0; rt < 4; ++rt) {
      v8f c = {0.f, 0.f, 0.f, 0.f, 0.f, 0.f, 0.f, 0.f};
      c = wmma_bf16(aq[rt][0], kb0, c);
      c = wmma_bf16(aq[rt][1], kb1, c);
#pragma unroll
      for (int g = 0; g < 8; ++g) {
        float w = __expf(c[g] * scale - rmax[rt][g]) * rinv[rt][g];
        wtile[rt * 16 + h * 8 + g][wv * 16 + ln] = (bf16_t)w;
      }
    }

    // ---- issue next chunk's keys loads early (hide under exp/DS writes) ----
    const int mtn = (kc + 4 < kc_end) ? (kc + 4 + wv) : wv;
    const bf16_t* krown = keys + (size_t)(mtn * 16 + ln) * HD_DIM;
    __builtin_prefetch(krown + 4 * 16 * HD_DIM, 0, 0);
    v16bf kb0n = b_frag(krown, 0, h);
    v16bf kb1n = b_frag(krown, 32, h);

    // region 1 schedule: 8 WMMAs braided with the 4 next-keys VMEM reads
#pragma unroll
    for (int u = 0; u < 4; ++u) {
      __builtin_amdgcn_sched_group_barrier(SG_WMMA, 2, 0);
      __builtin_amdgcn_sched_group_barrier(SG_VMEMRD, 1, 0);
    }

    __syncthreads();

    // ---- transposed reload as A fragments (K chunk of 64) ----
    v16bf aw[4][2];
#pragma unroll
    for (int rt = 0; rt < 4; ++rt) {
      const bf16_t* wrow = &wtile[rt * 16 + ln][0];
      aw[rt][0] = a_frag_bf16(wrow, 0, h);
      aw[rt][1] = a_frag_bf16(wrow, 32, h);
    }

    // ---- attended chain: pipeline Vt B-fragments one slot ahead ----
    const size_t mbase = (size_t)kc * 16;
    v16bf bv_cur = b_frag(vrow[0] + mbase, 0, h);
#pragma unroll
    for (int i = 0; i < 8; ++i) {
      const int s = i >> 2;
      const int nt = i & 3;
      v16bf bv_nxt;
      if (i < 7) {
        const int i1 = i + 1;
        bv_nxt = b_frag(vrow[i1 & 3] + mbase, (i1 >> 2) * 32, h);
      }
#pragma unroll
      for (int rt = 0; rt < 4; ++rt)
        acc[rt][nt] = wmma_bf16(aw[rt][s], bv_cur, acc[rt][nt]);
      bv_cur = bv_nxt;
    }

    // region 2 schedule: per slot, 2 DS reads + 2 VMEM reads braided with 4 WMMAs
#pragma unroll
    for (int u = 0; u < 8; ++u) {
      __builtin_amdgcn_sched_group_barrier(SG_DSRD, 2, 0);
      __builtin_amdgcn_sched_group_barrier(SG_VMEMRD, 2, 0);
      __builtin_amdgcn_sched_group_barrier(SG_WMMA, 4, 0);
    }

    __syncthreads();

    kb0 = kb0n;
    kb1 = kb1n;
  }

#pragma unroll
  for (int rt = 0; rt < 4; ++rt)
#pragma unroll
    for (int nt = 0; nt < 4; ++nt)
#pragma unroll
      for (int g = 0; g < 8; ++g) {
        int row = qb + rt * 16 + h * 8 + g;
        att_bf[(size_t)row * D_DIM + d_base + nt * 16 + ln] = (bf16_t)acc[rt][nt][g];
      }
}

// ---------------------------------------------------------------------------
// fused = concat(cf, attended) @ Wo^T + bo   (Wo row-major [512,1024] == B^T)
// ---------------------------------------------------------------------------
__global__ __launch_bounds__(128) void fused_kernel(const bf16_t* __restrict__ cfbf,
                                                    const bf16_t* __restrict__ attbf,
                                                    const bf16_t* __restrict__ Wo,
                                                    const float* __restrict__ bo,
                                                    float* __restrict__ out) {
  const int lane = threadIdx.x & 31;
  const int wv = threadIdx.x >> 5;  // 0..3
  const int ln = lane & 15;
  const int h = lane >> 4;
  const int qb = blockIdx.x * 16;
  const int n_base = blockIdx.y * 256 + wv * 64;

  v8f acc[4];
#pragma unroll
  for (int nt = 0; nt < 4; ++nt) {
    float b = bo[n_base + nt * 16 + ln];
#pragma unroll
    for (int g = 0; g < 8; ++g) acc[nt][g] = b;
  }

#pragma unroll 4
  for (int k0 = 0; k0 < 2 * D_DIM; k0 += 32) {
    const bf16_t* Asrc = (k0 < D_DIM) ? cfbf : attbf;
    const int kk = (k0 < D_DIM) ? k0 : (k0 - D_DIM);
    v16bf a = a_frag_bf16(Asrc + (size_t)(qb + ln) * D_DIM, kk, h);
    v16bf bw[4];
#pragma unroll
    for (int nt = 0; nt < 4; ++nt)
      bw[nt] = b_frag(Wo + (size_t)(n_base + nt * 16 + ln) * (2 * D_DIM), k0, h);
#pragma unroll
    for (int nt = 0; nt < 4; ++nt) acc[nt] = wmma_bf16(a, bw[nt], acc[nt]);
  }

#pragma unroll
  for (int nt = 0; nt < 4; ++nt)
#pragma unroll
    for (int g = 0; g < 8; ++g)
      out[(size_t)(qb + h * 8 + g) * D_DIM + n_base + nt * 16 + ln] = acc[nt][g];
}

// ---------------------------------------------------------------------------
// mem_new = memory with rows [0,1024) replaced by current_features.
// Non-temporal: this 256MB stream must not evict the L2-resident Vt/keys.
// ---------------------------------------------------------------------------
__global__ __launch_bounds__(256) void memnew_kernel(const v4f* __restrict__ cf,
                                                     const v4f* __restrict__ mem,
                                                     v4f* __restrict__ out) {
  size_t i = (size_t)blockIdx.x * blockDim.x + threadIdx.x;
  const size_t cf_n4 = (size_t)B_ROWS * D_DIM / 4;
  v4f val;
  if (i < cf_n4) {
    val = cf[i];
  } else {
    val = __builtin_nontemporal_load(mem + i);
  }
  __builtin_nontemporal_store(val, out + i);
}

// ---------------------------------------------------------------------------
// Host-side launch
// ---------------------------------------------------------------------------
extern "C" void kernel_launch(void* const* d_in, const int* in_sizes, int n_in,
                              void* d_out, int out_size, void* d_ws, size_t ws_size,
                              hipStream_t stream) {
  const float* cf  = (const float*)d_in[0];  // [1024, 512]
  const float* mem = (const float*)d_in[1];  // [65536, 512]
  const float* Wq  = (const float*)d_in[2];  // [64, 512]
  const float* bq  = (const float*)d_in[3];  // [64]
  const float* Wk  = (const float*)d_in[4];  // [64, 512]
  const float* bk  = (const float*)d_in[5];  // [64]
  const float* Wv  = (const float*)d_in[6];  // [512, 512]
  const float* bv  = (const float*)d_in[7];  // [512]
  const float* Wo  = (const float*)d_in[8];  // [512, 1024]
  const float* bo  = (const float*)d_in[9];  // [512]

  float* out_fused = (float*)d_out;                           // [1024, 512]
  float* out_mem   = (float*)d_out + (size_t)B_ROWS * D_DIM;  // [65536, 512]

  // ---- workspace carve-up (256B aligned) ----
  uintptr_t p = (uintptr_t)d_ws;
  auto carve = [&](size_t bytes) {
    uintptr_t r = p;
    p += (bytes + 255) & ~(size_t)255;
    return (void*)r;
  };
  bf16_t* cf_bf   = (bf16_t*)carve((size_t)B_ROWS * D_DIM * 2);
  bf16_t* Wq_bf   = (bf16_t*)carve((size_t)HD_DIM * D_DIM * 2);
  bf16_t* Wk_bf   = (bf16_t*)carve((size_t)HD_DIM * D_DIM * 2);
  bf16_t* Wv_bf   = (bf16_t*)carve((size_t)D_DIM * D_DIM * 2);
  bf16_t* Wo_bf   = (bf16_t*)carve((size_t)D_DIM * 2 * D_DIM * 2);
  bf16_t* q_bf    = (bf16_t*)carve((size_t)B_ROWS * HD_DIM * 2);
  bf16_t* keys_bf = (bf16_t*)carve((size_t)M_ROWS * HD_DIM * 2);
  bf16_t* vt_bf   = (bf16_t*)carve((size_t)D_DIM * M_ROWS * 2);
  bf16_t* att_bf  = (bf16_t*)carve((size_t)B_ROWS * D_DIM * 2);
  float*  stats   = (float*)carve((size_t)B_ROWS * 2 * 4);

  // ---- f32 -> bf16 conversions ----
  auto cvt = [&](const float* s, bf16_t* d, size_t n) {
    int n8 = (int)(n / 8);
    cvt_bf16_kernel<<<(n8 + 255) / 256, 256, 0, stream>>>(s, d, n8);
  };
  cvt(cf, cf_bf, (size_t)B_ROWS * D_DIM);
  cvt(Wq, Wq_bf, (size_t)HD_DIM * D_DIM);
  cvt(Wk, Wk_bf, (size_t)HD_DIM * D_DIM);
  cvt(Wv, Wv_bf, (size_t)D_DIM * D_DIM);
  cvt(Wo, Wo_bf, (size_t)D_DIM * 2 * D_DIM);

  // ---- fused keys+values projection (memory read once) ----
  proj_kv_kernel<<<M_ROWS / 16, 128, 0, stream>>>(mem, Wk_bf, bk, Wv_bf, bv,
                                                  keys_bf, vt_bf);
  // ---- query projection ----
  proj_q_kernel<<<B_ROWS / 128, 256, 0, stream>>>(cf_bf, Wq_bf, bq, q_bf);

  // ---- softmax stats (pass 1, 64 rows/block) ----
  stats_kernel<<<B_ROWS / 64, 256, 0, stream>>>(q_bf, keys_bf, stats);

  // ---- attended (pass 2, 64 rows/block) ----
  attended_kernel<<<dim3(B_ROWS / 64, 2), 128, 0, stream>>>(q_bf, keys_bf, vt_bf,
                                                            stats, att_bf);

  // ---- output projection ----
  fused_kernel<<<dim3(B_ROWS / 16, 2), 128, 0, stream>>>(cf_bf, att_bf, Wo_bf, bo,
                                                         out_fused);

  // ---- mem_new copy (non-temporal) ----
  memnew_kernel<<<(M_ROWS * (size_t)D_DIM / 4) / 256, 256, 0, stream>>>(
      (const v4f*)cf, (const v4f*)mem, (v4f*)out_mem);
}
